// SlotAttention_82222853915094
// MI455X (gfx1250) — compile-verified, matrix operand order checked by reading, not run on hardware
//
#include <hip/hip_runtime.h>
#include <hip/hip_bf16.h>
#include <stdint.h>

// ---------------------------------------------------------------------------
// Slot Attention fused pipeline for MI455X (gfx1250, wave32, WMMA).
//
// B=32, N=8192, D=512, NS=8 slots (padded to 16 for WMMA), 3 iterations.
// x and s are stored as bf16 hi/lo splits; every dot product is computed as
// hi*hi + hi*lo + lo*hi with v_wmma_f32_16x16x32_bf16 (f32 accumulate), giving
// near-f32 accuracy at matrix-pipe speed.
//
// Traffic: read inputs (536MB) + write split x (536MB) + 3x stream x (1.6GB)
// ~= 2.7GB  ->  ~115us at 23.3 TB/s. Phase-2 re-reads of x are L2 hits.
// ---------------------------------------------------------------------------

typedef __bf16 bf16;
typedef __attribute__((ext_vector_type(16))) __bf16 v16bf;
typedef __attribute__((ext_vector_type(8)))  float  v8f;

#define B_     32
#define N_     8192
#define D_     512
#define NSLOT  8
#define SPAD   16          // slot dim padded to 16 for 16x16 WMMA tiles
#define ITERS  3
#define JT     256         // j rows per block in attn kernel (8 waves x 32)
#define SCALE  0.04419417382415922f   // 512^-0.5
#define EPS_   1e-8f
#define LNEPS  1e-5f

struct FragBF {
  union {
    v16bf    v;
    uint32_t u[8];
    uint4    q[2];
  };
};

// dword built from two 16-bit loads -> encourages global_load_d16_b16 /
// global_load_d16_hi_b16 codegen (no shift/or VALU).
union U32H {
  uint32_t u;
  uint16_t h[2];
};

__device__ __forceinline__ uint16_t f2bf_bits(float f) {
  bf16 h = (bf16)f;                       // RNE
  return __builtin_bit_cast(uint16_t, h);
}
__device__ __forceinline__ float bfbits2f(uint16_t b) {
  bf16 h = __builtin_bit_cast(bf16, b);
  return (float)h;
}

// ---------------------------------------------------------------------------
// Kernel 1: LayerNorm of inputs -> x_hi/x_lo (bf16 split) + x_sq per row.
// One 256-thread block per row (2 elements per thread).
// ---------------------------------------------------------------------------
__global__ void ln_x_kernel(const float* __restrict__ in,
                            const float* __restrict__ w,
                            const float* __restrict__ bia,
                            uint16_t* __restrict__ xh,
                            uint16_t* __restrict__ xl,
                            float* __restrict__ xsq) {
  const int row = blockIdx.x;            // 0 .. B*N-1
  const int t   = threadIdx.x;           // 0 .. 255
  const float* p = in + (size_t)row * D_;
  float e0 = p[t], e1 = p[t + 256];
  float s1 = e0 + e1;
  float s2 = e0 * e0 + e1 * e1;

  __shared__ float red[2][8];
  for (int m = 16; m; m >>= 1) { s1 += __shfl_xor(s1, m); s2 += __shfl_xor(s2, m); }
  const int wv = t >> 5, ln = t & 31;
  if (ln == 0) { red[0][wv] = s1; red[1][wv] = s2; }
  __syncthreads();
  if (t == 0) {
    float a = 0.f, b2 = 0.f;
    for (int i = 0; i < 8; i++) { a += red[0][i]; b2 += red[1][i]; }
    red[0][0] = a; red[1][0] = b2;
  }
  __syncthreads();
  s1 = red[0][0]; s2 = red[1][0];
  const float mu  = s1 * (1.0f / D_);
  const float var = s2 * (1.0f / D_) - mu * mu;
  const float rs  = rsqrtf(var + LNEPS);
  const float y0 = (e0 - mu) * rs * w[t]       + bia[t];
  const float y1 = (e1 - mu) * rs * w[t + 256] + bia[t + 256];

  float q = y0 * y0 + y1 * y1;
  for (int m = 16; m; m >>= 1) q += __shfl_xor(q, m);
  __syncthreads();
  if (ln == 0) red[0][wv] = q;
  __syncthreads();
  if (t == 0) {
    float a = 0.f;
    for (int i = 0; i < 8; i++) a += red[0][i];
    xsq[row] = a;
  }

  const uint16_t h0 = f2bf_bits(y0);
  const uint16_t h1 = f2bf_bits(y1);
  const uint16_t l0 = f2bf_bits(y0 - bfbits2f(h0));
  const uint16_t l1 = f2bf_bits(y1 - bfbits2f(h1));
  const size_t base = (size_t)row * D_;
  xh[base + t] = h0;  xh[base + t + 256] = h1;
  xl[base + t] = l0;  xl[base + t + 256] = l1;
}

// ---------------------------------------------------------------------------
// Kernel 2: slot init: slots = mu + exp(logsigma) * noise   [B,8,D]
// ---------------------------------------------------------------------------
__global__ void init_slots_kernel(const float* __restrict__ noise,
                                  const float* __restrict__ mu,
                                  const float* __restrict__ lsig,
                                  float* __restrict__ slots) {
  const int i = blockIdx.x * 256 + threadIdx.x;
  if (i < B_ * NSLOT * D_) {
    const int d = i & (D_ - 1);
    slots[i] = mu[d] + __expf(lsig[d]) * noise[i];
  }
}

// ---------------------------------------------------------------------------
// Kernel 3: LayerNorm of slots -> s_hi/s_lo (padded [B,16,D], rows 8..15 = 0)
// + s_sq, and zero the num/den accumulators for the upcoming iteration.
// One block per padded row.
// ---------------------------------------------------------------------------
__global__ void ln_slots_kernel(const float* __restrict__ slots,
                                const float* __restrict__ w,
                                const float* __restrict__ bia,
                                uint16_t* __restrict__ sh,
                                uint16_t* __restrict__ sl,
                                float* __restrict__ ssq,
                                float* __restrict__ numw,
                                float* __restrict__ denw) {
  const int row  = blockIdx.x;          // 0 .. B*16-1
  const int b    = row >> 4;
  const int slot = row & 15;
  const int t    = threadIdx.x;
  const size_t pbase = (size_t)row * D_;

  if (t == 0) denw[row] = 0.0f;

  if (slot >= NSLOT) {                  // padding rows -> zeros
    sh[pbase + t] = 0; sh[pbase + t + 256] = 0;
    sl[pbase + t] = 0; sl[pbase + t + 256] = 0;
    if (t == 1) ssq[row] = 0.0f;
    return;
  }

  const size_t nbase = (size_t)(b * NSLOT + slot) * D_;
  numw[nbase + t] = 0.0f;               // zero accumulators
  numw[nbase + t + 256] = 0.0f;

  const float* p = slots + nbase;
  float e0 = p[t], e1 = p[t + 256];
  float s1 = e0 + e1, s2 = e0 * e0 + e1 * e1;

  __shared__ float red[2][8];
  for (int m = 16; m; m >>= 1) { s1 += __shfl_xor(s1, m); s2 += __shfl_xor(s2, m); }
  const int wv = t >> 5, ln = t & 31;
  if (ln == 0) { red[0][wv] = s1; red[1][wv] = s2; }
  __syncthreads();
  if (t == 0) {
    float a = 0.f, b2 = 0.f;
    for (int i = 0; i < 8; i++) { a += red[0][i]; b2 += red[1][i]; }
    red[0][0] = a; red[1][0] = b2;
  }
  __syncthreads();
  s1 = red[0][0]; s2 = red[1][0];
  const float mu  = s1 * (1.0f / D_);
  const float var = s2 * (1.0f / D_) - mu * mu;
  const float rs  = rsqrtf(var + LNEPS);
  const float y0 = (e0 - mu) * rs * w[t]       + bia[t];
  const float y1 = (e1 - mu) * rs * w[t + 256] + bia[t + 256];

  float q = y0 * y0 + y1 * y1;
  for (int m = 16; m; m >>= 1) q += __shfl_xor(q, m);
  __syncthreads();
  if (ln == 0) red[0][wv] = q;
  __syncthreads();
  if (t == 0) {
    float a = 0.f;
    for (int i = 0; i < 8; i++) a += red[0][i];
    ssq[row] = a;
  }

  const uint16_t h0 = f2bf_bits(y0);
  const uint16_t h1 = f2bf_bits(y1);
  const uint16_t l0 = f2bf_bits(y0 - bfbits2f(h0));
  const uint16_t l1 = f2bf_bits(y1 - bfbits2f(h1));
  sh[pbase + t] = h0;  sh[pbase + t + 256] = h1;
  sl[pbase + t] = l0;  sl[pbase + t + 256] = l1;
}

// ---------------------------------------------------------------------------
// Kernel 4: fused attention iteration. Grid (N/JT, B), 256 threads = 8 waves.
//
// Phase 1 (per wave, its own 32-j chunk):
//   dots[j,slot] via WMMA: A = x tile (M=j, K), B = s^T (K, N=slot).
//   Fragment loads are straight b128 loads thanks to the documented layouts
//   (A/B Vr holds elements  kb*32 + 8*half + {0..7} | {16..23}).
//   dots tile: lane -> slot = lane&15, VGPR r -> j = r + 8*(lane>=16).
//   Softmax over slots = shfl_xor(1,2,4) inside aligned 8-lane groups
//   (padding slots live in their own groups and are masked to 0).
//   attn is hi/lo-split, packed per j-pair and stored to LDS already in
//   B-fragment dword order.
// Phase 2 (per wave, its own 64-wide d slice, all 8 j-chunks):
//   num^T[d,slot] += x^T (A, d16/d16_hi gathers from L2-hot rows)
//                    x attn (B, ds_read_b32), accumulated in 32 VGPRs,
//   then merged with global_atomic_add_f32.
// ---------------------------------------------------------------------------
__global__ __launch_bounds__(256)
void attn_iter_kernel(const uint16_t* __restrict__ xh,
                      const uint16_t* __restrict__ xl,
                      const float* __restrict__ xsq,
                      const uint16_t* __restrict__ sh,
                      const uint16_t* __restrict__ sl,
                      const float* __restrict__ ssq,
                      float* __restrict__ numw,
                      float* __restrict__ denw) {
  __shared__ uint32_t pkh[8][16][16];   // packed bf16 attn pairs (hi part)
  __shared__ uint32_t pkl[8][16][16];   // packed bf16 attn pairs (lo part)

  const int b      = blockIdx.y;
  const int jblock = blockIdx.x * JT;
  const int tid    = threadIdx.x;
  const int w      = tid >> 5;          // wave id 0..7
  const int lane   = tid & 31;
  const int n16    = lane & 15;
  const int half   = lane >> 4;

  const uint16_t* xhB = xh + (size_t)b * N_ * D_;
  const uint16_t* xlB = xl + (size_t)b * N_ * D_;
  const uint16_t* shB = sh + (size_t)b * SPAD * D_;
  const uint16_t* slB = sl + (size_t)b * SPAD * D_;

  // ---------------- Phase 1: dots + softmax for this wave's 32-j chunk -----
  const int jc = w;
  const int j0 = jblock + jc * 32;

  const float ssq_n = ssq[b * SPAD + n16];

  v8f zf = {};
  v8f dacc0 = zf, dacc1 = zf;

  const uint16_t* sRowH = shB + (size_t)n16 * D_;
  const uint16_t* sRowL = slB + (size_t)n16 * D_;
  const uint16_t* xRowH0 = xhB + (size_t)(j0 + n16) * D_;
  const uint16_t* xRowL0 = xlB + (size_t)(j0 + n16) * D_;
  const uint16_t* xRowH1 = xRowH0 + (size_t)16 * D_;
  const uint16_t* xRowL1 = xRowL0 + (size_t)16 * D_;

  for (int kb = 0; kb < 16; kb++) {
    const int ke = kb * 32 + half * 8;
    FragBF bh, bl;
    bh.q[0] = *(const uint4*)(sRowH + ke);
    bh.q[1] = *(const uint4*)(sRowH + ke + 16);
    bl.q[0] = *(const uint4*)(sRowL + ke);
    bl.q[1] = *(const uint4*)(sRowL + ke + 16);

    FragBF ah, al;
    ah.q[0] = *(const uint4*)(xRowH0 + ke);
    ah.q[1] = *(const uint4*)(xRowH0 + ke + 16);
    al.q[0] = *(const uint4*)(xRowL0 + ke);
    al.q[1] = *(const uint4*)(xRowL0 + ke + 16);
    dacc0 = __builtin_amdgcn_wmma_f32_16x16x32_bf16(false, ah.v, false, bh.v, (short)0, dacc0, false, false);
    dacc0 = __builtin_amdgcn_wmma_f32_16x16x32_bf16(false, ah.v, false, bl.v, (short)0, dacc0, false, false);
    dacc0 = __builtin_amdgcn_wmma_f32_16x16x32_bf16(false, al.v, false, bh.v, (short)0, dacc0, false, false);

    ah.q[0] = *(const uint4*)(xRowH1 + ke);
    ah.q[1] = *(const uint4*)(xRowH1 + ke + 16);
    al.q[0] = *(const uint4*)(xRowL1 + ke);
    al.q[1] = *(const uint4*)(xRowL1 + ke + 16);
    dacc1 = __builtin_amdgcn_wmma_f32_16x16x32_bf16(false, ah.v, false, bh.v, (short)0, dacc1, false, false);
    dacc1 = __builtin_amdgcn_wmma_f32_16x16x32_bf16(false, ah.v, false, bl.v, (short)0, dacc1, false, false);
    dacc1 = __builtin_amdgcn_wmma_f32_16x16x32_bf16(false, al.v, false, bh.v, (short)0, dacc1, false, false);
  }

  const bool realSlot = (n16 < NSLOT);
  float denp = 0.0f;

  for (int t = 0; t < 2; t++) {
    const v8f dd = (t == 0) ? dacc0 : dacc1;
    float a[8];
    for (int r = 0; r < 8; r++) {
      const int j = j0 + t * 16 + r + half * 8;
      const float xs = xsq[(size_t)b * N_ + j];
      float dv = ssq_n + xs - 2.0f * dd[r];
      a[r] = fmaxf(dv, 0.0f) * SCALE;
    }
    for (int r = 0; r < 8; r++) {
      float v = a[r];
      float m = v;
      m = fmaxf(m, __shfl_xor(m, 1));
      m = fmaxf(m, __shfl_xor(m, 2));
      m = fmaxf(m, __shfl_xor(m, 4));
      float p = __expf(v - m);
      float s = p;
      s += __shfl_xor(s, 1);
      s += __shfl_xor(s, 2);
      s += __shfl_xor(s, 4);
      float at = p / s + EPS_;
      at = realSlot ? at : 0.0f;     // padding-slot columns -> 0
      a[r] = at;
      denp += at;
    }
    // pack attn pairs (j even/odd = VGPR r, r+1 of same lane) into LDS,
    // already in B-fragment dword layout.
    for (int r = 0; r < 8; r += 2) {
      const uint16_t h0 = f2bf_bits(a[r]);
      const uint16_t h1 = f2bf_bits(a[r + 1]);
      const uint16_t l0 = f2bf_bits(a[r]     - bfbits2f(h0));
      const uint16_t l1 = f2bf_bits(a[r + 1] - bfbits2f(h1));
      const int jp = t * 8 + half * 4 + (r >> 1);
      pkh[jc][jp][n16] = (uint32_t)h0 | ((uint32_t)h1 << 16);
      pkl[jc][jp][n16] = (uint32_t)l0 | ((uint32_t)l1 << 16);
    }
  }

  // per-slot denominator partial: combine lane halves, add from lanes 0..7
  denp += __shfl_xor(denp, 16);
  if (lane < NSLOT) atomicAdd(&denw[b * SPAD + lane], denp);

  __syncthreads();

  // ---------------- Phase 2: num^T[d,slot] += x^T * attn -------------------
  const int dbase = w * 64;
  v8f acc0 = zf, acc1 = zf, acc2 = zf, acc3 = zf;

  for (int c = 0; c < 8; c++) {
    FragBF bh, bl;
    for (int r = 0; r < 8; r++) {
      const int jp = 4 * half + (r & 3) + 8 * (r >> 2);
      bh.u[r] = pkh[c][jp][n16];
      bl.u[r] = pkl[c][jp][n16];
    }
    const int jcbase = jblock + c * 32;
    for (int dt = 0; dt < 4; dt++) {
      const int d = dbase + dt * 16 + n16;
      FragBF ah, al;
      for (int r = 0; r < 8; r++) {
        const int jj = 8 * half + 2 * (r & 3) + 16 * (r >> 2);
        const size_t off0 = (size_t)(jcbase + jj) * D_ + d;
        const size_t off1 = off0 + D_;
        U32H th, tl;                       // -> global_load_d16_b16 / _hi
        th.h[0] = xhB[off0];
        th.h[1] = xhB[off1];
        tl.h[0] = xlB[off0];
        tl.h[1] = xlB[off1];
        ah.u[r] = th.u;
        al.u[r] = tl.u;
      }
      v8f acc = (dt == 0) ? acc0 : (dt == 1) ? acc1 : (dt == 2) ? acc2 : acc3;
      acc = __builtin_amdgcn_wmma_f32_16x16x32_bf16(false, ah.v, false, bh.v, (short)0, acc, false, false);
      acc = __builtin_amdgcn_wmma_f32_16x16x32_bf16(false, ah.v, false, bl.v, (short)0, acc, false, false);
      acc = __builtin_amdgcn_wmma_f32_16x16x32_bf16(false, al.v, false, bh.v, (short)0, acc, false, false);
      if (dt == 0) acc0 = acc; else if (dt == 1) acc1 = acc; else if (dt == 2) acc2 = acc; else acc3 = acc;
    }
  }

  // merge partial results: D2 tile lane -> slot = n16, d = dbase+dt*16+r+8*half
  if (realSlot) {
    float* numB = numw + ((size_t)b * NSLOT + n16) * D_;
    for (int dt = 0; dt < 4; dt++) {
      const v8f acc = (dt == 0) ? acc0 : (dt == 1) ? acc1 : (dt == 2) ? acc2 : acc3;
      for (int r = 0; r < 8; r++) {
        const int d = dbase + dt * 16 + r + 8 * half;
        atomicAdd(&numB[d], acc[r]);
      }
    }
  }
}

// ---------------------------------------------------------------------------
// Kernel 5: slots = num / den ; also writes final output on last iteration.
// ---------------------------------------------------------------------------
__global__ void update_slots_kernel(const float* __restrict__ numw,
                                    const float* __restrict__ denw,
                                    float* __restrict__ slots,
                                    float* __restrict__ out) {
  const int row  = blockIdx.x;          // 0 .. B*8-1
  const int b    = row >> 3;
  const int slot = row & 7;
  const int t    = threadIdx.x;
  const float inv = 1.0f / denw[b * SPAD + slot];
  const size_t base = (size_t)row * D_;
  const float v0 = numw[base + t] * inv;
  const float v1 = numw[base + t + 256] * inv;
  slots[base + t] = v0;
  slots[base + t + 256] = v1;
  if (out) {
    out[base + t] = v0;
    out[base + t + 256] = v1;
  }
}

// ---------------------------------------------------------------------------
// Host launcher
// ---------------------------------------------------------------------------
extern "C" void kernel_launch(void* const* d_in, const int* in_sizes, int n_in,
                              void* d_out, int out_size, void* d_ws, size_t ws_size,
                              hipStream_t stream) {
  (void)in_sizes; (void)n_in; (void)out_size; (void)ws_size;

  const float* inputs         = (const float*)d_in[0];
  const float* slot_noise     = (const float*)d_in[1];
  const float* slots_mu       = (const float*)d_in[2];
  const float* slots_logsigma = (const float*)d_in[3];
  const float* ln_in_w        = (const float*)d_in[4];
  const float* ln_in_b        = (const float*)d_in[5];
  const float* ln_slots_w     = (const float*)d_in[6];
  const float* ln_slots_b     = (const float*)d_in[7];

  char* ws = (char*)d_ws;
  size_t off = 0;
  auto carve = [&](size_t bytes) -> char* {
    char* p = ws + off;
    off += (bytes + 255) & ~(size_t)255;
    return p;
  };

  uint16_t* xh   = (uint16_t*)carve((size_t)B_ * N_ * D_ * 2);
  uint16_t* xl   = (uint16_t*)carve((size_t)B_ * N_ * D_ * 2);
  float*    xsq  = (float*)   carve((size_t)B_ * N_ * 4);
  uint16_t* sh   = (uint16_t*)carve((size_t)B_ * SPAD * D_ * 2);
  uint16_t* sl   = (uint16_t*)carve((size_t)B_ * SPAD * D_ * 2);
  float*    ssq  = (float*)   carve((size_t)B_ * SPAD * 4);
  float*    slots= (float*)   carve((size_t)B_ * NSLOT * D_ * 4);
  float*    numw = (float*)   carve((size_t)B_ * NSLOT * D_ * 4);
  float*    denw = (float*)   carve((size_t)B_ * SPAD * 4);

  ln_x_kernel<<<dim3(B_ * N_), dim3(256), 0, stream>>>(
      inputs, ln_in_w, ln_in_b, xh, xl, xsq);

  init_slots_kernel<<<dim3((B_ * NSLOT * D_ + 255) / 256), dim3(256), 0, stream>>>(
      slot_noise, slots_mu, slots_logsigma, slots);

  for (int it = 0; it < ITERS; it++) {
    ln_slots_kernel<<<dim3(B_ * SPAD), dim3(256), 0, stream>>>(
        slots, ln_slots_w, ln_slots_b, sh, sl, ssq, numw, denw);

    attn_iter_kernel<<<dim3(N_ / JT, B_), dim3(256), 0, stream>>>(
        xh, xl, xsq, sh, sl, ssq, numw, denw);

    update_slots_kernel<<<dim3(B_ * NSLOT), dim3(256), 0, stream>>>(
        numw, denw, slots, (it == ITERS - 1) ? (float*)d_out : nullptr);
  }
}
